// HomoMoleculeGNN_GPS_18013092839581
// MI455X (gfx1250) — matrix operand order repaired and checked by reading, not compiled, not run on hardware
//
#include <hip/hip_runtime.h>

// ---------------- problem constants ----------------
constexpr int GG      = 512;           // graphs
constexpr int NPGc    = 128;           // nodes per graph
constexpr int CC      = 256;           // channels
constexpr int NN      = GG * NPGc;     // 65536 nodes
constexpr int DEGc    = 4;
constexpr int EE      = NN * DEGc;     // 262144 edges
constexpr int HEADSc  = 4;
constexpr int INCH    = 232;
constexpr int NWALKS  = 20;
#define BN_EPS 1e-5f

// ---------------- WMMA types ----------------
typedef __attribute__((ext_vector_type(16))) __bf16 v16bf;
typedef __attribute__((ext_vector_type(8)))  float  v8f;
typedef __attribute__((ext_vector_type(2)))  float  v2f;

union BFrag { v16bf v; unsigned u[8]; };

__device__ inline unsigned short f2bf(float x) {
  unsigned u = __builtin_bit_cast(unsigned, x);
  u += 0x7FFFu + ((u >> 16) & 1u);          // RNE
  return (unsigned short)(u >> 16);
}
__device__ inline unsigned pack2(unsigned short lo, unsigned short hi) {
  return (unsigned)lo | ((unsigned)hi << 16);
}

// CDNA5 async copy: global -> LDS, 16 bytes per lane, tracked by ASYNCcnt.
__device__ inline void async_ld_b128(unsigned ldsaddr, const unsigned short* gaddr) {
  asm volatile("global_load_async_to_lds_b128 %0, %1, off"
               :: "v"(ldsaddr), "v"(gaddr) : "memory");
}
__device__ inline void wait_async0() {
  asm volatile("s_wait_asynccnt 0" ::: "memory");
}

// A-fragment 16x32 bf16 (rows of a row-major matrix, K contiguous).
__device__ inline v16bf frag_a_rows(const unsigned short* p, int ld) {
  int lane = threadIdx.x & 31, half = lane >> 4, r = lane & 15;
  const unsigned short* rp = p + (size_t)r * ld + half * 16;
  BFrag f;
#pragma unroll
  for (int j = 0; j < 8; ++j) {
    int kb = 2 * j;
    f.u[j] = *(const unsigned*)(rp + kb);
  }
  return f.v;
}
// B-fragment 32x16 bf16 where B[k,n] = W[n,k] (W rows row-major).
__device__ inline v16bf frag_b_rows(const unsigned short* p, int ld) {
  int lane = threadIdx.x & 31, half = lane >> 4, r = lane & 15;
  const unsigned short* rp = p + (size_t)r * ld + half * 16;
  BFrag f;
#pragma unroll
  for (int j = 0; j < 8; ++j) f.u[j] = *(const unsigned*)(rp + 2 * j);
  return f.v;
}
// B-fragment 32x16 bf16 from a truly row-major B (element (k,n) at p[k*ld+n]).
__device__ inline v16bf frag_b_col(const unsigned short* p, int ld) {
  int lane = threadIdx.x & 31, half = lane >> 4, r = lane & 15;
  const unsigned short* cp = p + r + (size_t)(half * 16) * ld;
  BFrag f;
#pragma unroll
  for (int j = 0; j < 8; ++j)
    f.u[j] = pack2(cp[(size_t)(2 * j) * ld], cp[(size_t)(2 * j + 1) * ld]);
  return f.v;
}
__device__ inline v8f wmma_bf(v16bf a, v16bf b, v8f c) {
  return __builtin_amdgcn_wmma_f32_16x16x32_bf16(false, a, false, b, (short)0, c,
                                                 false, false);
}

// ------- generic NT GEMM: out = A(bf16)[MxK] * W(bf16)[NxK]^T (+bias,+addend,relu) ----
// out (f32) and out_bf (bf16) are each optional.
__global__ __launch_bounds__(256) void gemm_bf16_nt(
    const unsigned short* __restrict__ A, int lda,
    const unsigned short* __restrict__ W, int ldw,
    const float* __restrict__ bias, const float* __restrict__ addend,
    float* __restrict__ out, unsigned short* __restrict__ out_bf, int ldo,
    int M, int N, int K, int do_relu) {
  int wave = (blockIdx.x << 3) + (threadIdx.x >> 5);
  int nt64 = N >> 6;
  int m0 = (wave / nt64) << 5;
  int n0 = (wave % nt64) << 6;
  if (m0 >= M) return;
  v8f z = {0.f, 0.f, 0.f, 0.f, 0.f, 0.f, 0.f, 0.f};
  v8f acc[2][4];
#pragma unroll
  for (int a = 0; a < 2; ++a)
#pragma unroll
    for (int b = 0; b < 4; ++b) acc[a][b] = z;
  for (int k0 = 0; k0 < K; k0 += 32) {
    v16bf a0 = frag_a_rows(A + (size_t)m0 * lda + k0, lda);
    v16bf a1 = frag_a_rows(A + (size_t)(m0 + 16) * lda + k0, lda);
#pragma unroll
    for (int t = 0; t < 4; ++t) {
      v16bf b = frag_b_rows(W + (size_t)(n0 + 16 * t) * ldw + k0, ldw);
      acc[0][t] = wmma_bf(a0, b, acc[0][t]);
      acc[1][t] = wmma_bf(a1, b, acc[1][t]);
    }
  }
  int lane = threadIdx.x & 31, half = lane >> 4, r = lane & 15;
#pragma unroll
  for (int mt = 0; mt < 2; ++mt)
#pragma unroll
    for (int t = 0; t < 4; ++t)
#pragma unroll
      for (int v = 0; v < 8; ++v) {
        int row = m0 + mt * 16 + v + 8 * half;
        int col = n0 + 16 * t + r;
        float val = acc[mt][t][v];
        if (bias) val += bias[col];
        if (addend) val += addend[(size_t)row * ldo + col];
        if (do_relu) val = fmaxf(val, 0.f);
        if (out) out[(size_t)row * ldo + col] = val;
        if (out_bf) out_bf[(size_t)row * ldo + col] = f2bf(val);
      }
}

// ---------------- RWSE: diag(A_rw^k), k=1..20, per graph (f32 WMMA) ----------------
__global__ __launch_bounds__(256) void rwse_kernel(const int* __restrict__ eidx,
                                                   float* __restrict__ pe) {
  extern __shared__ char smem[];
  float* Amat = (float*)smem;              // 128 x 132 (padded)
  float* Cur = Amat + 128 * 132;           // 128 x 132
  int* deg = (int*)(Cur + 128 * 132);      // 128
  const int g = blockIdx.x;
  const int tid = threadIdx.x;
  for (int i = tid; i < 128 * 132; i += 256) Amat[i] = 0.f;
  if (tid < 128) deg[tid] = 0;
  __syncthreads();
  const int* rowi = eidx;
  const int* coli = eidx + EE;
  const int ebase = g * (NPGc * DEGc);
  for (int e = tid; e < NPGc * DEGc; e += 256)
    atomicAdd(&deg[rowi[ebase + e] & 127], 1);
  __syncthreads();
  for (int e = tid; e < NPGc * DEGc; e += 256) {
    int s = rowi[ebase + e] & 127;
    int d = coli[ebase + e] & 127;
    float val = 1.f / (float)max(deg[s], 1);
    atomicAdd(&Amat[s * 132 + d], val);
  }
  __syncthreads();
  for (int i = tid; i < 128 * 132; i += 256) Cur[i] = Amat[i];
  __syncthreads();
  if (tid < 128) pe[(size_t)(g * 128 + tid) * NWALKS + 0] = Cur[tid * 132 + tid];
  const int w = tid >> 5, lane = tid & 31, half = lane >> 4, r = lane & 15;
  const int m = w * 16 + r;
  for (int s = 1; s < NWALKS; ++s) {
    v8f z = {0.f, 0.f, 0.f, 0.f, 0.f, 0.f, 0.f, 0.f};
    v8f acc[8];
#pragma unroll
    for (int t = 0; t < 8; ++t) acc[t] = z;
    for (int k0 = 0; k0 < 128; k0 += 4) {
      v2f a;
      a.x = Cur[m * 132 + k0 + half * 2];
      a.y = Cur[m * 132 + k0 + half * 2 + 1];
#pragma unroll
      for (int tn = 0; tn < 8; ++tn) {
        int n = tn * 16 + r;
        v2f b;
        b.x = Amat[(k0 + half * 2) * 132 + n];
        b.y = Amat[(k0 + half * 2 + 1) * 132 + n];
        acc[tn] = __builtin_amdgcn_wmma_f32_16x16x4_f32(false, a, false, b, (short)0,
                                                        acc[tn], false, false);
      }
    }
    __syncthreads();
#pragma unroll
    for (int tn = 0; tn < 8; ++tn)
#pragma unroll
      for (int v = 0; v < 8; ++v)
        Cur[(w * 16 + v + 8 * half) * 132 + tn * 16 + r] = acc[tn][v];
    __syncthreads();
    if (tid < 128) pe[(size_t)(g * 128 + tid) * NWALKS + s] = Cur[tid * 132 + tid];
  }
}

// ---------------- BatchNorm column statistics ----------------
__global__ __launch_bounds__(256) void col_stats(const float* __restrict__ x, int Mrows,
                                                 int ld, float* __restrict__ mean,
                                                 float* __restrict__ rstd) {
  __shared__ float s1[256], s2[256];
  int c = blockIdx.x;
  float a = 0.f, b = 0.f;
  for (int i = threadIdx.x; i < Mrows; i += 256) {
    float v = x[(size_t)i * ld + c];
    a += v;
    b += v * v;
  }
  s1[threadIdx.x] = a;
  s2[threadIdx.x] = b;
  __syncthreads();
  for (int s = 128; s > 0; s >>= 1) {
    if (threadIdx.x < (unsigned)s) {
      s1[threadIdx.x] += s1[threadIdx.x + s];
      s2[threadIdx.x] += s2[threadIdx.x + s];
    }
    __syncthreads();
  }
  if (threadIdx.x == 0) {
    float mu = s1[0] / (float)Mrows;
    float var = s2[0] / (float)Mrows - mu * mu;
    mean[c] = mu;
    rstd[c] = rsqrtf(fmaxf(var, 0.f) + BN_EPS);
  }
}

__global__ void bn_apply(const float* __restrict__ in, float* __restrict__ out,
                         unsigned short* __restrict__ out_bf,
                         const float* __restrict__ g, const float* __restrict__ b,
                         const float* __restrict__ mean, const float* __restrict__ rstd,
                         int ld, size_t total) {
  size_t i = (size_t)blockIdx.x * 256 + threadIdx.x;
  if (i >= total) return;
  int c = (int)(i % (size_t)ld);
  float v = g[c] * (in[i] - mean[c]) * rstd[c] + b[c];
  out[i] = v;
  if (out_bf) out_bf[i] = f2bf(v);
}

__global__ void cvt_bf16(const float* __restrict__ in, unsigned short* __restrict__ out,
                         size_t n) {
  size_t i = (size_t)blockIdx.x * 256 + threadIdx.x;
  if (i < n) out[i] = f2bf(in[i]);
}

__global__ void add2(const float* __restrict__ a, const float* __restrict__ b,
                     float* __restrict__ out, unsigned short* __restrict__ out_bf,
                     size_t n) {
  size_t i = (size_t)blockIdx.x * 256 + threadIdx.x;
  if (i < n) {
    float v = a[i] + b[i];
    out[i] = v;
    out_bf[i] = f2bf(v);
  }
}

// ---------------- input feature assembly ----------------
__global__ __launch_bounds__(256) void build_h0(
    const float* __restrict__ x, const int* __restrict__ ntypes,
    const float* __restrict__ ntype_emb, const float* __restrict__ peraw,
    const float* __restrict__ pe_g, const float* __restrict__ pe_b,
    const float* __restrict__ pe_mean, const float* __restrict__ pe_rstd,
    const float* __restrict__ pe_lin_w, const float* __restrict__ pe_lin_b,
    float* __restrict__ H, unsigned short* __restrict__ Hbf) {
  int n = blockIdx.x, c = threadIdx.x;
  float v;
  if (c < 16)
    v = ntype_emb[ntypes[n] * 16 + c];
  else if (c < 16 + INCH)
    v = x[(size_t)n * INCH + (c - 16)];
  else {
    int d = c - (16 + INCH);
    float acc = pe_lin_b[d];
    for (int k = 0; k < NWALKS; ++k) {
      float t = pe_g[k] * (peraw[(size_t)n * NWALKS + k] - pe_mean[k]) * pe_rstd[k] +
                pe_b[k];
      acc += pe_lin_w[d * NWALKS + k] * t;
    }
    v = acc;
  }
  H[(size_t)n * CC + c] = v;
  Hbf[(size_t)n * CC + c] = f2bf(v);
}

__global__ void build_ebf(const float* __restrict__ eattr, const int* __restrict__ etypes,
                          const float* __restrict__ etype_emb,
                          unsigned short* __restrict__ EBF) {
  size_t i = (size_t)blockIdx.x * 256 + threadIdx.x;  // < E*32
  int e = (int)(i >> 5), c = (int)(i & 31);
  float v = (c < 16) ? etype_emb[etypes[e] * 16 + c]
                     : eattr[(size_t)e * 16 + (c - 16)];
  EBF[i] = f2bf(v);
}

// ---------------- GINE gather-aggregate (circulant incoming edges) ----------------
__global__ __launch_bounds__(256) void gine_agg(const float* __restrict__ H,
                                                const float* __restrict__ el,
                                                unsigned short* __restrict__ out_bf) {
  int n = blockIdx.x, c = threadIdx.x;
  int g = n >> 7, j = n & 127;
  float acc = H[(size_t)n * CC + c];
#pragma unroll
  for (int o = 1; o <= DEGc; ++o) {
    int i = (j - o) & 127;
    int src = (g << 7) + i;
    size_t e = (size_t)g * 512 + (size_t)(o - 1) * 128 + i;
    float m = H[(size_t)src * CC + c] + el[e * CC + c];
    acc += fmaxf(m, 0.f);
  }
  out_bf[(size_t)n * CC + c] = f2bf(acc);
}

// ---------------- per-(graph,head) attention (async LDS staging) ----------------
constexpr int QKV_PITCH = 72;   // bf16, padded (row = 9x16B -> 16B aligned chunks)
constexpr int S_PITCH = 130;    // f32, padded
constexpr int P_PITCH = 136;    // bf16, padded
constexpr int ATTN_SMEM = 3 * 128 * QKV_PITCH * 2 + 128 * S_PITCH * 4 + 128 * P_PITCH * 2;
constexpr int RWSE_SMEM = 2 * 128 * 132 * 4 + 128 * 4;

__global__ __launch_bounds__(256) void attn_kernel(const unsigned short* __restrict__ qkv,
                                                   unsigned short* __restrict__ o) {
  extern __shared__ char smem[];
  unsigned short* Qs = (unsigned short*)smem;
  unsigned short* Ks = Qs + 128 * QKV_PITCH;
  unsigned short* Vs = Ks + 128 * QKV_PITCH;
  float* Ss = (float*)(Vs + 128 * QKV_PITCH);
  unsigned short* Pb = (unsigned short*)(Ss + 128 * S_PITCH);
  const int g = blockIdx.x >> 2, h = blockIdx.x & 3;
  const int tid = threadIdx.x;
  const unsigned lds0 = (unsigned)(size_t)smem;
  // Async-stage Q/K/V (bf16) into padded LDS: 3 mats * 128 rows * 8 chunks of 16B.
#pragma unroll
  for (int t = 0; t < 12; ++t) {
    int idx = t * 256 + tid;                 // [0, 3072)
    int mat = idx >> 10;                     // 0:q 1:k 2:v
    int rem = idx & 1023;
    int row = rem >> 3, seg = rem & 7;
    const unsigned short* gp =
        qkv + (size_t)(g * 128 + row) * 768 + mat * 256 + h * 64 + seg * 8;
    unsigned lp = lds0 + (unsigned)(mat * 128 * QKV_PITCH * 2 +
                                    (row * QKV_PITCH + seg * 8) * 2);
    async_ld_b128(lp, gp);
  }
  wait_async0();
  __syncthreads();
  const int w = tid >> 5, lane = tid & 31, half = lane >> 4, r = lane & 15;
  v8f z = {0.f, 0.f, 0.f, 0.f, 0.f, 0.f, 0.f, 0.f};
  // S = Q K^T * 1/sqrt(64)
#pragma unroll
  for (int tn = 0; tn < 8; ++tn) {
    v8f acc = z;
#pragma unroll
    for (int k0 = 0; k0 < 64; k0 += 32) {
      v16bf a = frag_a_rows(Qs + w * 16 * QKV_PITCH + k0, QKV_PITCH);
      v16bf b = frag_b_rows(Ks + tn * 16 * QKV_PITCH + k0, QKV_PITCH);
      acc = wmma_bf(a, b, acc);
    }
#pragma unroll
    for (int v = 0; v < 8; ++v)
      Ss[(w * 16 + v + 8 * half) * S_PITCH + tn * 16 + r] = acc[v] * 0.125f;
  }
  __syncthreads();
  if (tid < 128) {
    float* rowp = Ss + tid * S_PITCH;
    float mx = -1e30f;
    for (int i = 0; i < 128; ++i) mx = fmaxf(mx, rowp[i]);
    float sum = 0.f;
    for (int i = 0; i < 128; ++i) {
      float e = __expf(rowp[i] - mx);
      rowp[i] = e;
      sum += e;
    }
    float inv = 1.f / sum;
    for (int i = 0; i < 128; ++i) Pb[tid * P_PITCH + i] = f2bf(rowp[i] * inv);
  }
  __syncthreads();
  // O = P V  (bf16 out, feeds out-projection GEMM)
#pragma unroll
  for (int tn = 0; tn < 4; ++tn) {
    v8f acc = z;
#pragma unroll
    for (int k0 = 0; k0 < 128; k0 += 32) {
      v16bf a = frag_a_rows(Pb + w * 16 * P_PITCH + k0, P_PITCH);
      v16bf b = frag_b_col(Vs + (size_t)k0 * QKV_PITCH + tn * 16, QKV_PITCH);
      acc = wmma_bf(a, b, acc);
    }
#pragma unroll
    for (int v = 0; v < 8; ++v) {
      int rr = w * 16 + v + 8 * half, cc = tn * 16 + r;
      o[(size_t)(g * 128 + rr) * CC + h * 64 + cc] = f2bf(acc[v]);
    }
  }
}

// ---------------- host orchestration ----------------
extern "C" void kernel_launch(void* const* d_in, const int* in_sizes, int n_in,
                              void* d_out, int out_size, void* d_ws, size_t ws_size,
                              hipStream_t stream) {
  (void)in_sizes; (void)n_in; (void)out_size; (void)ws_size;
  // Flatten order: top-level dict insertion order, params pytree sorted by key.
  const float* x_in = (const float*)d_in[0];
  const int* eidx = (const int*)d_in[1];
  const int* ntypes = (const int*)d_in[2];
  const int* etypes = (const int*)d_in[3];
  const float* eattr = (const float*)d_in[4];
  const float* etype_emb = (const float*)d_in[45];
  const float* ntype_emb = (const float*)d_in[46];
  const float* pe_b = (const float*)d_in[47];
  const float* pe_g = (const float*)d_in[48];
  const float* pe_lin_b = (const float*)d_in[49];
  const float* pe_lin_w = (const float*)d_in[50];

  size_t off = 0;
  auto alloc = [&](size_t bytes) -> void* {
    off = (off + 255) & ~(size_t)255;
    void* p = (char*)d_ws + off;
    off += bytes;
    return p;
  };
  float* H = (float*)alloc((size_t)NN * CC * 4);
  float* R1 = (float*)alloc((size_t)NN * CC * 4);
  float* R2 = (float*)alloc((size_t)NN * CC * 4);
  float* R3 = (float*)alloc((size_t)NN * CC * 4);
  float* BIG = (float*)alloc((size_t)EE * CC * 4);               // el (f32)
  unsigned short* BF1 = (unsigned short*)alloc((size_t)NN * 768 * 2);
  unsigned short* BF2 = (unsigned short*)alloc((size_t)NN * 512 * 2);
  unsigned short* EBF = (unsigned short*)alloc((size_t)EE * 32 * 2);
  float* PERAW = (float*)alloc((size_t)NN * NWALKS * 4);
  float* STAT = (float*)alloc(4096 * 4);
  unsigned short* WBF = (unsigned short*)alloc((size_t)2 * 663552 * 2);

  hipFuncSetAttribute((const void*)rwse_kernel,
                      hipFuncAttributeMaxDynamicSharedMemorySize, RWSE_SMEM);
  hipFuncSetAttribute((const void*)attn_kernel,
                      hipFuncAttributeMaxDynamicSharedMemorySize, ATTN_SMEM);

  float* MEAN = STAT;
  float* RSTD = STAT + 512;

  // ---- stage 0: RWSE + PE BN + feature concat + edge features ----
  rwse_kernel<<<GG, 256, RWSE_SMEM, stream>>>(eidx, PERAW);
  col_stats<<<NWALKS, 256, 0, stream>>>(PERAW, NN, NWALKS, MEAN, RSTD);
  build_h0<<<NN, 256, 0, stream>>>(x_in, ntypes, ntype_emb, PERAW, pe_g, pe_b, MEAN,
                                   RSTD, pe_lin_w, pe_lin_b, H, BF2);
  build_ebf<<<(EE * 32) / 256, 256, 0, stream>>>(eattr, etypes, etype_emb, EBF);

  // ---- convert weights to bf16 (both layers) ----
  static const int WOFF[7] = {0, 8192, 73728, 139264, 335872, 401408, 532480};
  static const int WCNT[7] = {8192, 65536, 65536, 196608, 65536, 131072, 131072};
  static const int WIDX[7] = {5, 18, 19, 1, 3, 8, 9};  // lin_e,nn1,nn2,ain,aout,m1,m2
  for (int l = 0; l < 2; ++l)
    for (int wi = 0; wi < 7; ++wi) {
      const float* wp = (const float*)d_in[5 + 20 * l + WIDX[wi]];
      cvt_bf16<<<(WCNT[wi] + 255) / 256, 256, 0, stream>>>(
          wp, WBF + (size_t)l * 663552 + WOFF[wi], (size_t)WCNT[wi]);
    }

  auto gemm = [&](const unsigned short* A, int lda, const unsigned short* W, int ldw,
                  const float* bias, const float* addend, float* out,
                  unsigned short* out_bf, int ldo, int M, int N, int K, int relu) {
    int blocks = (M / 32) * (N / 64) / 8;
    gemm_bf16_nt<<<blocks, 256, 0, stream>>>(A, lda, W, ldw, bias, addend, out, out_bf,
                                             ldo, M, N, K, relu);
  };
  const size_t NC = (size_t)NN * CC;
  const int NCB = (int)(NC / 256);

  for (int l = 0; l < 2; ++l) {
    int b = 5 + 20 * l;
    const float* attn_in_b = (const float*)d_in[b + 0];
    const float* attn_out_b = (const float*)d_in[b + 2];
    const float* lin_e_b = (const float*)d_in[b + 4];
    const float* mlp_b1 = (const float*)d_in[b + 6];
    const float* mlp_b2 = (const float*)d_in[b + 7];
    const float* n1_b = (const float*)d_in[b + 10];
    const float* n1_g = (const float*)d_in[b + 11];
    const float* n2_b = (const float*)d_in[b + 12];
    const float* n2_g = (const float*)d_in[b + 13];
    const float* n3_b = (const float*)d_in[b + 14];
    const float* n3_g = (const float*)d_in[b + 15];
    const float* nn_b1 = (const float*)d_in[b + 16];
    const float* nn_b2 = (const float*)d_in[b + 17];
    const unsigned short* Wl = WBF + (size_t)l * 663552;

    // ---- attention branch (uses BF2 = h_bf from previous stage) ----
    // qkv (bf16 only) = h @ attn_in_w^T + b   [N x 768]
    gemm(BF2, CC, Wl + WOFF[3], CC, attn_in_b, nullptr, nullptr, BF1, 768,
         NN, 768, CC, 0);
    // per-(graph,head) attention; o (bf16) -> BF2 (h_bf consumed above)
    attn_kernel<<<GG * HEADSc, 256, ATTN_SMEM, stream>>>(BF1, BF2);
    // h2pre = o @ attn_out_w^T + b + h
    gemm(BF2, CC, Wl + WOFF[4], CC, attn_out_b, H, R3, nullptr, CC, NN, CC, CC, 0);
    col_stats<<<CC, 256, 0, stream>>>(R3, NN, CC, MEAN, RSTD);
    bn_apply<<<NCB, 256, 0, stream>>>(R3, R3, nullptr, n2_g, n2_b, MEAN, RSTD, CC, NC);

    // ---- GINE branch ----
    // el = ein @ lin_e_w^T + b   [E x 256] (f32)
    gemm(EBF, 32, Wl + WOFF[0], 32, lin_e_b, nullptr, BIG, nullptr, CC, EE, CC, 32, 0);
    // hsum (bf16) = h + sum_o relu(h[src] + el)
    gine_agg<<<NN, 256, 0, stream>>>(H, BIG, BF1);
    // t1 (bf16 only) = relu(hsum @ nn_w1^T + b1)
    gemm(BF1, CC, Wl + WOFF[1], CC, nn_b1, nullptr, nullptr, BF2, CC, NN, CC, CC, 1);
    // h1pre = t1 @ nn_w2^T + b2 + h
    gemm(BF2, CC, Wl + WOFF[2], CC, nn_b2, H, R1, nullptr, CC, NN, CC, CC, 0);
    col_stats<<<CC, 256, 0, stream>>>(R1, NN, CC, MEAN, RSTD);
    bn_apply<<<NCB, 256, 0, stream>>>(R1, R1, nullptr, n1_g, n1_b, MEAN, RSTD, CC, NC);

    // ---- combine + MLP ----
    // out = h1 + h2 (f32 R2 + bf16 BF1)
    add2<<<NCB, 256, 0, stream>>>(R1, R3, R2, BF1, NC);
    // mlp hidden (bf16 only) = relu(out @ mlp_w1^T + b1)   [N x 512]
    gemm(BF1, CC, Wl + WOFF[5], CC, mlp_b1, nullptr, nullptr, BF2, 512, NN, 512, CC, 1);
    // mlppre = hidden @ mlp_w2^T + b2 + out
    gemm(BF2, 512, Wl + WOFF[6], 512, mlp_b2, R2, R1, nullptr, CC, NN, CC, 512, 0);
    col_stats<<<CC, 256, 0, stream>>>(R1, NN, CC, MEAN, RSTD);
    float* tgt = (l == 1) ? (float*)d_out : H;
    // bn3 -> next h (f32) + h_bf (BF2) for next layer
    bn_apply<<<NCB, 256, 0, stream>>>(R1, tgt, BF2, n3_g, n3_b, MEAN, RSTD, CC, NC);
  }
}
